// TransBlock_4578435138058
// MI455X (gfx1250) — matrix operand order repaired and checked by reading, not compiled, not run on hardware
//
#include <hip/hip_runtime.h>
#include <math.h>

// CDNA5 (gfx1250) wave32 WMMA types
typedef __attribute__((ext_vector_type(16))) _Float16 v16h;
typedef __attribute__((ext_vector_type(8)))  _Float16 v8h;
typedef __attribute__((ext_vector_type(8)))  float    v8f;

#define GEMM_WAVES 8   // 256 threads/block = 8 wave32

// ---------------------------------------------------------------------------
// CDNA5 async global->LDS helpers (ASYNCcnt-tracked DMA, ISA ch.10 async ops).
// Dynamic LDS starts at byte offset 0, so we pass raw LDS byte offsets as the
// VDST operand (the LDS address is that VGPR plus the instruction offset).
// ---------------------------------------------------------------------------
__device__ __forceinline__ void async_copy_b128(unsigned lds_byte_off,
                                                const void* gaddr) {
    asm volatile("global_load_async_to_lds_b128 %0, %1, off"
                 :: "v"(lds_byte_off), "v"(gaddr) : "memory");
}
__device__ __forceinline__ void async_wait0() {
    asm volatile("s_wait_asynccnt 0" ::: "memory");
}

// ---------------------------------------------------------------------------
// Weight pack: B[K,N] f32 row-major  ->  Bp[N,K] f16 (per-column K-contiguous,
// matching the B-fragment loads in gemm_wmma_f16: each lane's 16 halves are K-consecutive)
// ---------------------------------------------------------------------------
__global__ __launch_bounds__(256) void pack_bT_f16(
    const float* __restrict__ B, _Float16* __restrict__ Bp, int K, int N)
{
    const int idx = blockIdx.x * 256 + threadIdx.x;
    if (idx >= K * N) return;
    const int k = idx / N;
    const int n = idx - k * N;
    Bp[(size_t)n * K + k] = (_Float16)B[idx];
}

// ---------------------------------------------------------------------------
// GEMM: C[M,N] = act(A[M,K] @ B[K,N] + bias[N]); B pre-packed as Bp[N,K] f16.
// One wave computes a 16x32 C patch: two v_wmma_f32_16x16x32_f16 per K-step,
// sharing one A fragment. A: 4x float4 loads + v_cvt_pk; B: 2x b128 per tile.
// ---------------------------------------------------------------------------
template <bool GELU>
__global__ __launch_bounds__(256) void gemm_wmma_f16(
    const float* __restrict__ A, const _Float16* __restrict__ Bp,
    const float* __restrict__ bias, float* __restrict__ C,
    int M, int N, int Kdim)
{
    const int lane = threadIdx.x & 31;
    const int wave = threadIdx.x >> 5;
    const int tilesN2  = N >> 5;                    // 32-wide column blocks
    const int numTiles = (M >> 4) * tilesN2;
    const int tile = blockIdx.x * GEMM_WAVES + wave;
    if (tile >= numTiles) return;
    const int tm  = tile / tilesN2;
    const int tn2 = tile - tm * tilesN2;

    const int l15 = lane & 15;
    const bool hi = (lane >= 16);
    // 16-bit A layout: 16 halves K-consecutive starting at koffA (matches B)
    const int koffA = hi ? 16 : 0;
    // 16-bit B layout (same order as A): 16 halves K-consecutive starting at koffB
    const int koffB = hi ? 16 : 0;

    const float4*   Arow = (const float4*)(A + (size_t)(tm * 16 + l15) * Kdim);
    const _Float16* B0   = Bp + (size_t)(tn2 * 32 + l15) * Kdim;
    const _Float16* B1   = B0 + (size_t)16 * Kdim;

    v8f acc0 = {}, acc1 = {};
    for (int k0 = 0; k0 < Kdim; k0 += 32) {
        if (k0 + 32 < Kdim) {
            __builtin_prefetch((const float*)Arow + k0 + 32, 0, 1);  // global_prefetch_b8
            __builtin_prefetch(B0 + k0 + 32, 0, 1);
            __builtin_prefetch(B1 + k0 + 32, 0, 1);
        }
        const int ka = (k0 + koffA) >> 2;           // float4 index, 16B aligned
        const float4 a0 = Arow[ka],     a1 = Arow[ka + 1];
        const float4 a2 = Arow[ka + 2], a3 = Arow[ka + 3];   // +8 floats
        v16h a;
        a[0]=(_Float16)a0.x; a[1]=(_Float16)a0.y; a[2]=(_Float16)a0.z; a[3]=(_Float16)a0.w;
        a[4]=(_Float16)a1.x; a[5]=(_Float16)a1.y; a[6]=(_Float16)a1.z; a[7]=(_Float16)a1.w;
        a[8]=(_Float16)a2.x; a[9]=(_Float16)a2.y; a[10]=(_Float16)a2.z; a[11]=(_Float16)a2.w;
        a[12]=(_Float16)a3.x; a[13]=(_Float16)a3.y; a[14]=(_Float16)a3.z; a[15]=(_Float16)a3.w;

        {   // first N-subtile
            const v8h* bp = (const v8h*)(B0 + k0 + koffB);   // 32B aligned
            const v8h blo = bp[0], bhi = bp[1];
            v16h b;
#pragma unroll
            for (int h = 0; h < 8; ++h) { b[h] = blo[h]; b[h + 8] = bhi[h]; }
            acc0 = __builtin_amdgcn_wmma_f32_16x16x32_f16(
                false, a, false, b, (short)0, acc0, false, false);
        }
        {   // second N-subtile (A fragment reused)
            const v8h* bp = (const v8h*)(B1 + k0 + koffB);
            const v8h blo = bp[0], bhi = bp[1];
            v16h b;
#pragma unroll
            for (int h = 0; h < 8; ++h) { b[h] = blo[h]; b[h + 8] = bhi[h]; }
            acc1 = __builtin_amdgcn_wmma_f32_16x16x32_f16(
                false, a, false, b, (short)0, acc1, false, false);
        }
    }

    // D layout: VGPR r -> row tm*16 + r + (hi?8:0), col = lane&15 within tile
    const int mbase = tm * 16 + (hi ? 8 : 0);
    const int n0 = tn2 * 32 + l15;
    const int n1 = n0 + 16;
    const float bn0 = bias[n0], bn1 = bias[n1];
#pragma unroll
    for (int r = 0; r < 8; ++r) {
        float v0 = acc0[r] + bn0;
        float v1 = acc1[r] + bn1;
        if (GELU) {
            v0 = 0.5f * v0 * (1.0f + erff(v0 * 0.70710678118654752f));
            v1 = 0.5f * v1 * (1.0f + erff(v1 * 0.70710678118654752f));
        }
        C[(size_t)(mbase + r) * N + n0] = v0;
        C[(size_t)(mbase + r) * N + n1] = v1;
    }
}

// ---------------------------------------------------------------------------
// Neighborhood attention (7x7, NATTEN border clamping).
// One block per pixel; the block async-DMAs its 49 neighbor rows' k|v slab
// (49 x 256 f32 = 49KB) into LDS (ASYNCcnt), then one wave per head computes
// scores with wave32 butterfly reductions and the weighted sum from LDS.
// ---------------------------------------------------------------------------
__global__ __launch_bounds__(128) void na_attn7(
    const float* __restrict__ qkv,   // [M,384] : q|k|v
    float* __restrict__ out,         // [M,128]
    int Bsz, int H, int W)
{
    extern __shared__ float kv[];    // [49][256] : per-neighbor k(128)|v(128)

    const int pix  = blockIdx.x;
    const int head = threadIdx.x >> 5;
    const int lane = threadIdx.x & 31;
    const int HW = H * W;
    const int b  = pix / HW;
    const int rem = pix - b * HW;
    const int i  = rem / W;
    const int j  = rem - i * W;
    int is = i - 3; if (is < 0) is = 0; if (is > H - 7) is = H - 7;
    int js = j - 3; if (js < 0) js = 0; if (js > W - 7) js = W - 7;

    // Stage k|v of all 49 neighbors: 49*64 float4 chunks via async b128 DMA.
    for (int u = threadIdx.x; u < 49 * 64; u += 128) {
        const int n  = u >> 6;
        const int c4 = u & 63;                       // float4 index within row
        const int ni = is + n / 7;
        const int nj = js + n % 7;
        const float* src = qkv + (size_t)(b * HW + ni * W + nj) * 384 + 128 + c4 * 4;
        async_copy_b128((unsigned)((n * 256 + c4 * 4) * 4), src);
    }
    async_wait0();
    __syncthreads();

    const float scale = 0.1767766952966369f;         // 32^-0.5
    const int hc = head * 32 + lane;
    const float q = qkv[(size_t)pix * 384 + hc] * scale;

    float s[49];
#pragma unroll
    for (int n = 0; n < 49; ++n) {
        float p = q * kv[n * 256 + hc];
#pragma unroll
        for (int off = 16; off > 0; off >>= 1)
            p += __shfl_xor(p, off, 32);
        s[n] = p;                                    // all lanes hold the score
    }

    float m = s[0];
#pragma unroll
    for (int n = 1; n < 49; ++n) m = fmaxf(m, s[n]);
    float sum = 0.f;
#pragma unroll
    for (int n = 0; n < 49; ++n) { s[n] = __expf(s[n] - m); sum += s[n]; }
    const float inv = 1.0f / sum;

    float o = 0.f;
#pragma unroll
    for (int n = 0; n < 49; ++n)
        o += (s[n] * inv) * kv[n * 256 + 128 + hc];
    out[(size_t)pix * 128 + hc] = o;
}

// ---------------------------------------------------------------------------
// out = LayerNorm(y)*g + b + res, row length 128. One wave per row.
// ---------------------------------------------------------------------------
__global__ __launch_bounds__(256) void ln_residual128(
    const float* __restrict__ y, const float* __restrict__ g,
    const float* __restrict__ beta, const float* __restrict__ res,
    float* __restrict__ out, int M)
{
    const int lane = threadIdx.x & 31;
    const int wave = threadIdx.x >> 5;
    const int row  = blockIdx.x * 8 + wave;
    if (row >= M) return;

    const float4 v = ((const float4*)(y + (size_t)row * 128))[lane];
    float acc = v.x + v.y + v.z + v.w;
#pragma unroll
    for (int off = 16; off > 0; off >>= 1) acc += __shfl_xor(acc, off, 32);
    const float mean = acc * (1.0f / 128.0f);

    const float cx = v.x - mean, cy = v.y - mean, cz = v.z - mean, cw = v.w - mean;
    float vac = cx * cx + cy * cy + cz * cz + cw * cw;
#pragma unroll
    for (int off = 16; off > 0; off >>= 1) vac += __shfl_xor(vac, off, 32);
    const float rs = rsqrtf(vac * (1.0f / 128.0f) + 1e-5f);

    const float4 gv = ((const float4*)g)[lane];
    const float4 bv = ((const float4*)beta)[lane];
    const float4 rv = ((const float4*)(res + (size_t)row * 128))[lane];
    float4 o;
    o.x = cx * rs * gv.x + bv.x + rv.x;
    o.y = cy * rs * gv.y + bv.y + rv.y;
    o.z = cz * rs * gv.z + bv.z + rv.z;
    o.w = cw * rs * gv.w + bv.w + rv.w;
    ((float4*)(out + (size_t)row * 128))[lane] = o;
}

// ---------------------------------------------------------------------------
extern "C" void kernel_launch(void* const* d_in, const int* in_sizes, int n_in,
                              void* d_out, int out_size, void* d_ws, size_t ws_size,
                              hipStream_t stream)
{
    const float* x      = (const float*)d_in[0];
    const float* w_qkv  = (const float*)d_in[1];
    const float* b_qkv  = (const float*)d_in[2];
    const float* w_proj = (const float*)d_in[3];
    const float* b_proj = (const float*)d_in[4];
    const float* ln1_g  = (const float*)d_in[5];
    const float* ln1_b  = (const float*)d_in[6];
    const float* w_fc1  = (const float*)d_in[7];
    const float* b_fc1  = (const float*)d_in[8];
    const float* w_fc2  = (const float*)d_in[9];
    const float* b_fc2  = (const float*)d_in[10];
    const float* ln2_g  = (const float*)d_in[11];
    const float* ln2_b  = (const float*)d_in[12];

    const int B = 4, H = 56, W = 56, C = 128, MLP = 512;
    const int M = B * H * W;                         // 12544 rows (multiple of 16)

    // f32 scratch
    float* ws   = (float*)d_ws;
    float* qkv  = ws;                                // M*384
    float* attn = qkv  + (size_t)M * 384;            // M*128
    float* proj = attn + (size_t)M * 128;            // M*128
    float* x1   = proj + (size_t)M * 128;            // M*128
    float* h1   = x1   + (size_t)M * 128;            // M*512
    float* h2   = h1   + (size_t)M * 512;            // M*128
    // f16 packed weights [N,K]
    _Float16* wp      = (_Float16*)(h2 + (size_t)M * 128);
    _Float16* bp_qkv  = wp;                          // 384*128
    _Float16* bp_proj = bp_qkv  + 384 * 128;         // 128*128
    _Float16* bp_fc1  = bp_proj + 128 * 128;         // 512*128
    _Float16* bp_fc2  = bp_fc1  + 512 * 128;         // 128*512

    const dim3 blk(256);
    auto pblocks = [](int kn) { return (kn + 255) / 256; };
    auto gblocks = [](int m, int n) {
        return ((m >> 4) * (n >> 5) + GEMM_WAVES - 1) / GEMM_WAVES;
    };

    // 0) pack weights -> f16 [N,K]
    pack_bT_f16<<<pblocks(C * 3 * C), blk, 0, stream>>>(w_qkv,  bp_qkv,  C,   3 * C);
    pack_bT_f16<<<pblocks(C * C),     blk, 0, stream>>>(w_proj, bp_proj, C,   C);
    pack_bT_f16<<<pblocks(C * MLP),   blk, 0, stream>>>(w_fc1,  bp_fc1,  C,   MLP);
    pack_bT_f16<<<pblocks(MLP * C),   blk, 0, stream>>>(w_fc2,  bp_fc2,  MLP, C);

    // 1) QKV projection
    gemm_wmma_f16<false><<<gblocks(M, 3 * C), blk, 0, stream>>>(
        x, bp_qkv, b_qkv, qkv, M, 3 * C, C);

    // 2) 7x7 neighborhood attention (async LDS staging)
    na_attn7<<<M, 128, 49 * 256 * sizeof(float), stream>>>(qkv, attn, B, H, W);

    // 3) output projection
    gemm_wmma_f16<false><<<gblocks(M, C), blk, 0, stream>>>(
        attn, bp_proj, b_proj, proj, M, C, C);

    // 4) x1 = LN(proj)*g+b + x
    ln_residual128<<<(M + 7) / 8, blk, 0, stream>>>(proj, ln1_g, ln1_b, x, x1, M);

    // 5) FC1 + exact GELU
    gemm_wmma_f16<true><<<gblocks(M, MLP), blk, 0, stream>>>(
        x1, bp_fc1, b_fc1, h1, M, MLP, C);

    // 6) FC2
    gemm_wmma_f16<false><<<gblocks(M, C), blk, 0, stream>>>(
        h1, bp_fc2, b_fc2, h2, M, C, MLP);

    // 7) out = LN(h2)*g+b + x1
    ln_residual128<<<(M + 7) / 8, blk, 0, stream>>>(
        h2, ln2_g, ln2_b, x1, (float*)d_out, M);
}
